// Attention_84293028151648
// MI455X (gfx1250) — matrix operand order repaired and checked by reading, not compiled, not run on hardware
//
#include <hip/hip_runtime.h>
#include <hip/hip_bf16.h>
#include <math.h>

typedef __attribute__((ext_vector_type(16))) _Float16 v16h;
typedef __attribute__((ext_vector_type(8)))  _Float16 v8h;
typedef __attribute__((ext_vector_type(8)))  float    v8f;

#define BATCH 2
#define SEQ_Q 2048
#define SEQ_K 4096
#define DMODEL 1024
#define NHEAD 16
#define HDIM 64

// ---------------------------------------------------------------------------
// f32 -> f16 elementwise convert (activations)
// ---------------------------------------------------------------------------
__global__ void cvt_f32_to_f16_kernel(const float* __restrict__ in,
                                      _Float16* __restrict__ out, int n) {
  int i = blockIdx.x * blockDim.x + threadIdx.x;
  int stride = gridDim.x * blockDim.x;
  for (; i < n; i += stride) out[i] = (_Float16)in[i];
}

// ---------------------------------------------------------------------------
// Pack weight W[N,K] (row-major f32) into a WMMA f16 B-fragment layout using contiguous k order (not AMD's table order).
// For GEMM C = A @ W^T, the B fragment element for (lane, e) of tile
// (ntile, ktile) is  W[n, k]  with
//   n = ntile*16 + (lane & 15)
//   k = ktile*32 + (lane >> 4)*16 + e         (e = 0..15, contiguous in k)
// Stored as Bp[((ntile*(K/32)+ktile)*32 + lane)*16 + e]  -> one contiguous
// 32-byte chunk per lane per tile.
// ---------------------------------------------------------------------------
__global__ void pack_w_kernel(const float* __restrict__ W,
                              _Float16* __restrict__ Bp, int N, int K) {
  int t = blockIdx.x * blockDim.x + threadIdx.x;          // one (tile,lane) chunk
  int ktiles = K / 32;
  int total = (N / 16) * ktiles * 32;
  if (t >= total) return;
  int lane  = t & 31;
  int ktile = (t >> 5) % ktiles;
  int ntile = t / (32 * ktiles);
  int n     = ntile * 16 + (lane & 15);
  int kbase = ktile * 32 + (lane >> 4) * 16;
  const float* src = W + (size_t)n * K + kbase;           // contiguous 16 f32
  _Float16* dst = Bp + (size_t)t * 16;
#pragma unroll
  for (int e = 0; e < 16; ++e) dst[e] = (_Float16)src[e];
}

// ---------------------------------------------------------------------------
// Async-copy one K-stage of packed B (4 n-tiles x 1KB = 4KB) into LDS using
// the gfx1250 async global->LDS path (tracked by ASYNCcnt). 128 threads move
// 256 x 16B pieces -> 2 async b128 ops per thread (2 ASYNCcnt ticks per wave).
// ---------------------------------------------------------------------------
__device__ __forceinline__ void async_load_b_stage(const _Float16* __restrict__ Bp,
                                                   _Float16* buf,   // LDS, 2048 f16
                                                   int nt0, int ktiles, int kt) {
  const int i = threadIdx.x;                  // 0..127
#pragma unroll
  for (int p = 0; p < 2; ++p) {
    int P = i * 2 + p;                        // 16B piece id, 0..255
    int c = P >> 6;                           // which n-tile chunk (0..3)
    int inner = P & 63;                       // 16B piece within the 1KB chunk
    const _Float16* src =
        Bp + ((size_t)(nt0 + c) * ktiles + kt) * 512 + inner * 8;
    unsigned lds_addr = (unsigned)(uintptr_t)(buf + P * 8);   // LDS byte offset
    asm volatile("global_load_async_to_lds_b128 %0, %1, off"
                 :: "v"(lds_addr), "v"(src)
                 : "memory");
  }
}

// ---------------------------------------------------------------------------
// WMMA GEMM:  C[M,N] (f32) = A[M,K] (f16, row-major) @ W^T  + bias
// Block = 128 threads (4 waves), computes a 128x64 tile: each wave owns a
// 32x64 sub-tile (2 A fragments x 4 accumulators = 8 WMMA per K-step).
// B tiles are shared block-wide through a double-buffered LDS stage filled
// with async global->LDS copies and consumed via ds_load.
// ---------------------------------------------------------------------------
__global__ void __launch_bounds__(128)
gemm_wmma_kernel(const _Float16* __restrict__ A,
                 const _Float16* __restrict__ Bp,
                 const float* __restrict__ bias,
                 float* __restrict__ C,
                 int M, int N, int K) {
  __shared__ _Float16 smem[2][4 * 512];       // 2 stages x 4KB

  const int lane  = threadIdx.x & 31;
  const int wave  = threadIdx.x >> 5;
  const int mbase = (blockIdx.x * 4 + wave) * 2;   // first of 2 m-tiles
  const int nbase = blockIdx.y * 64;
  const int nt0   = nbase >> 4;
  const int mrow  = lane & 15;      // M index in A frag / N index in C frag
  const int khalf = lane >> 4;      // K-half held by this lane
  const int ktiles = K / 32;

  const _Float16* arow0 = A + (size_t)(mbase * 16 + mrow) * K;
  const _Float16* arow1 = arow0 + (size_t)16 * K;

  v8f acc[2][4];
#pragma unroll
  for (int s = 0; s < 2; ++s)
#pragma unroll
    for (int t = 0; t < 4; ++t) acc[s][t] = (v8f){};

  // Prologue: stage 0 in flight.
  async_load_b_stage(Bp, smem[0], nt0, ktiles, 0);

  for (int kt = 0; kt < ktiles; ++kt) {
    const _Float16* buf = smem[kt & 1];

    if (kt + 1 < ktiles) {
      async_load_b_stage(Bp, smem[(kt + 1) & 1], nt0, ktiles, kt + 1);
      // 2 newer ops outstanding; <=2 guarantees this stage's 2 ops landed.
      asm volatile("s_wait_asynccnt 0x2" ::: "memory");
    } else {
      asm volatile("s_wait_asynccnt 0x0" ::: "memory");
    }
    __syncthreads();                 // stage kt visible to all 4 waves

    const int kk = kt * 32;
    // A fragments (contiguous k order, matching packed B): lanes 0-15 hold K 0..15,
    // lanes 16-31 hold K 16..31 -> two contiguous 16B loads per tile.
    v8h a0lo = *(const v8h*)(arow0 + kk + khalf * 16);
    v8h a0hi = *(const v8h*)(arow0 + kk + khalf * 16 + 8);
    v8h a1lo = *(const v8h*)(arow1 + kk + khalf * 16);
    v8h a1hi = *(const v8h*)(arow1 + kk + khalf * 16 + 8);
    v16h a0, a1;
#pragma unroll
    for (int i = 0; i < 8; ++i) {
      a0[i] = a0lo[i]; a0[i + 8] = a0hi[i];
      a1[i] = a1lo[i]; a1[i + 8] = a1hi[i];
    }
    __builtin_prefetch(arow0 + kk + 32, 0, 1);   // global_prefetch_b8
    __builtin_prefetch(arow1 + kk + 32, 0, 1);

#pragma unroll
    for (int t = 0; t < 4; ++t) {
      v16h b = *(const v16h*)(buf + t * 512 + lane * 16);   // ds_load, 32B
      acc[0][t] = __builtin_amdgcn_wmma_f32_16x16x32_f16(
          false, a0, false, b, (short)0, acc[0][t], false, false);
      acc[1][t] = __builtin_amdgcn_wmma_f32_16x16x32_f16(
          false, a1, false, b, (short)0, acc[1][t], false, false);
    }
    __syncthreads();                 // all waves done reading before overwrite
  }

  // f32 C/D layout: VGPR r -> (M = khalf*8 + r, N = lane&15) within tile.
#pragma unroll
  for (int s = 0; s < 2; ++s) {
#pragma unroll
    for (int t = 0; t < 4; ++t) {
      int col = nbase + t * 16 + mrow;
      float bv = bias[col];
#pragma unroll
      for (int r = 0; r < 8; ++r) {
        int row = (mbase + s) * 16 + khalf * 8 + r;
        C[(size_t)row * N + col] = acc[s][t][r] + bv;
      }
    }
  }
}

// ---------------------------------------------------------------------------
// Local attention: mask allows query q to see exactly keys {2q, 2q+1}
// (SK/SQ == 2, remainder 0). Softmax over 2 elements; negligible FLOPs.
// One thread per (b, q, h). Output written as f16 for the final WMMA GEMM.
// ---------------------------------------------------------------------------
__global__ void attn_local_kernel(const float* __restrict__ Q,
                                  const float* __restrict__ Km,
                                  const float* __restrict__ Vm,
                                  _Float16* __restrict__ Ao) {
  int idx = blockIdx.x * blockDim.x + threadIdx.x;    // over BATCH*SEQ_Q*NHEAD
  int b = idx / (SEQ_Q * NHEAD);
  int r = idx % (SEQ_Q * NHEAD);
  int q = r / NHEAD;
  int h = r % NHEAD;

  const float* qv = Q + ((size_t)(b * SEQ_Q + q)) * DMODEL + h * HDIM;
  size_t krow = (size_t)(b * SEQ_K + 2 * q) * DMODEL + h * HDIM;
  const float* k0 = Km + krow;
  const float* k1 = k0 + DMODEL;
  const float* v0 = Vm + krow;
  const float* v1 = v0 + DMODEL;

  float s0 = 0.f, s1 = 0.f;
#pragma unroll 4
  for (int d = 0; d < HDIM; ++d) {
    float qd = qv[d];
    s0 += qd * k0[d];
    s1 += qd * k1[d];
  }
  const float scale = 0.125f;                         // 1/sqrt(64)
  s0 *= scale; s1 *= scale;
  float mx = fmaxf(s0, s1);
  float e0 = __expf(s0 - mx), e1 = __expf(s1 - mx);
  float inv = 1.f / (e0 + e1);
  float p0 = e0 * inv, p1 = e1 * inv;

  _Float16* dst = Ao + ((size_t)(b * SEQ_Q + q)) * DMODEL + h * HDIM;
#pragma unroll 4
  for (int d = 0; d < HDIM; ++d)
    dst[d] = (_Float16)(p0 * v0[d] + p1 * v1[d]);
}

// ---------------------------------------------------------------------------
extern "C" void kernel_launch(void* const* d_in, const int* in_sizes, int n_in,
                              void* d_out, int out_size, void* d_ws, size_t ws_size,
                              hipStream_t stream) {
  const float* x  = (const float*)d_in[0];
  const float* c  = (const float*)d_in[1];
  const float* Wq = (const float*)d_in[2];
  const float* bq = (const float*)d_in[3];
  const float* Wk = (const float*)d_in[4];
  const float* bk = (const float*)d_in[5];
  const float* Wv = (const float*)d_in[6];
  const float* bv = (const float*)d_in[7];
  const float* Wo = (const float*)d_in[8];
  const float* bo = (const float*)d_in[9];
  float* out = (float*)d_out;

  const int MQ = BATCH * SEQ_Q;   // 4096
  const int MK = BATCH * SEQ_K;   // 8192

  // Workspace carve-up (all offsets well aligned by construction).
  char* ws = (char*)d_ws;
  size_t off = 0;
  _Float16* xh  = (_Float16*)(ws + off); off += (size_t)MQ * DMODEL * 2;      //  8 MB
  _Float16* ch  = (_Float16*)(ws + off); off += (size_t)MK * DMODEL * 2;      // 16 MB
  _Float16* WqP = (_Float16*)(ws + off); off += (size_t)DMODEL * DMODEL * 2;  //  2 MB
  _Float16* WkP = (_Float16*)(ws + off); off += (size_t)DMODEL * DMODEL * 2;
  _Float16* WvP = (_Float16*)(ws + off); off += (size_t)DMODEL * DMODEL * 2;
  _Float16* WoP = (_Float16*)(ws + off); off += (size_t)DMODEL * DMODEL * 2;
  float*    Qf  = (float*)(ws + off);    off += (size_t)MQ * DMODEL * 4;      // 16 MB
  float*    Kf  = (float*)(ws + off);    off += (size_t)MK * DMODEL * 4;      // 32 MB
  float*    Vf  = (float*)(ws + off);    off += (size_t)MK * DMODEL * 4;      // 32 MB
  _Float16* Ao  = (_Float16*)(ws + off); off += (size_t)MQ * DMODEL * 2;      //  8 MB

  // 1) activations -> f16
  cvt_f32_to_f16_kernel<<<512, 256, 0, stream>>>(x, xh, MQ * DMODEL);
  cvt_f32_to_f16_kernel<<<512, 256, 0, stream>>>(c, ch, MK * DMODEL);

  // 2) pack weights into WMMA B layout ((D/16)*(D/32)*32 = 65536 lane-chunks)
  pack_w_kernel<<<256, 256, 0, stream>>>(Wq, WqP, DMODEL, DMODEL);
  pack_w_kernel<<<256, 256, 0, stream>>>(Wk, WkP, DMODEL, DMODEL);
  pack_w_kernel<<<256, 256, 0, stream>>>(Wv, WvP, DMODEL, DMODEL);
  pack_w_kernel<<<256, 256, 0, stream>>>(Wo, WoP, DMODEL, DMODEL);

  // 3) projections: grid = (M/128, N/64), 4 waves/block, 32x64 tile per wave
  dim3 blk(128);
  gemm_wmma_kernel<<<dim3(MQ / 128, DMODEL / 64), blk, 0, stream>>>(
      xh, WqP, bq, Qf, MQ, DMODEL, DMODEL);
  gemm_wmma_kernel<<<dim3(MK / 128, DMODEL / 64), blk, 0, stream>>>(
      ch, WkP, bk, Kf, MK, DMODEL, DMODEL);
  gemm_wmma_kernel<<<dim3(MK / 128, DMODEL / 64), blk, 0, stream>>>(
      ch, WvP, bv, Vf, MK, DMODEL, DMODEL);

  // 4) 2-key local attention -> f16 activations
  attn_local_kernel<<<(BATCH * SEQ_Q * NHEAD) / 256, 256, 0, stream>>>(
      Qf, Kf, Vf, Ao);

  // 5) output projection straight into d_out (f32)
  gemm_wmma_kernel<<<dim3(MQ / 128, DMODEL / 64), blk, 0, stream>>>(
      Ao, WoP, bo, out, MQ, DMODEL, DMODEL);
}
